// CausalSelfAttention_42176578846970
// MI455X (gfx1250) — matrix operand order repaired and checked by reading, not compiled, not run on hardware
//
#include <hip/hip_runtime.h>
#include <hip/hip_bf16.h>
#include <cstdint>

typedef _Float16 h16;
typedef __attribute__((ext_vector_type(16))) _Float16 v16h;
typedef __attribute__((ext_vector_type(8)))  _Float16 v8h;
typedef __attribute__((ext_vector_type(8)))  float    v8f;
typedef unsigned int u32x4 __attribute__((ext_vector_type(4)));
typedef int          i32x8 __attribute__((ext_vector_type(8)));
typedef int          i32x4 __attribute__((ext_vector_type(4)));

#define BDIM 2
#define TLEN 2048
#define EMB  2048
#define NH   16
#define HD   128
#define D2   256
#define H2G  8
#define WIN  1024
#define BT   (BDIM*TLEN)

#if defined(__has_builtin)
#if __has_builtin(__builtin_amdgcn_tensor_load_to_lds) && \
    __has_builtin(__builtin_amdgcn_s_wait_tensorcnt)
#define HAVE_TDM 1
#endif
#endif

#ifdef HAVE_TDM
#warning "CDNA5 TDM path: ENABLED (tensor_load_to_lds + s_wait_tensorcnt)"
#else
#warning "CDNA5 TDM path: DISABLED (cooperative global->LDS fallback)"
#endif

union AF { v16h v; v8h h[2]; };

__device__ __forceinline__ v8f wmma_f16(v16h a, v16h b, v8f c) {
  return __builtin_amdgcn_wmma_f32_16x16x32_f16(false, a, false, b, (short)0, c,
                                                false, false);
}

#ifdef HAVE_TDM
// Tensor-DMA a 2D tile (tile_x contiguous f16 elems x tile_y rows) into LDS.
// D# packing per CDNA5 ISA.
__device__ __forceinline__ void tdm_load_2d(const void* gaddr, void* lds,
                                            unsigned tile_x, unsigned tile_y,
                                            unsigned stride_elems,
                                            unsigned tensor_x, unsigned tensor_y) {
  unsigned long long ga = (unsigned long long)(uintptr_t)gaddr;
  unsigned la = (unsigned)(uintptr_t)lds;   // LDS aperture: low 32 bits = LDS offset
  u32x4 g0;
  g0[0] = 1u;                                             // count=1, user desc
  g0[1] = la;                                             // lds_addr
  g0[2] = (unsigned)(ga & 0xffffffffu);                   // global_addr[31:0]
  g0[3] = (unsigned)((ga >> 32) & 0x1ffffffu) | (2u << 30); // ga[56:32] | type=2
  i32x8 g1;
  g1[0] = (int)(1u << 16);                                // data_size=1 (2 bytes)
  g1[1] = (int)((tensor_x & 0xffffu) << 16);              // tensor_dim0[15:0]
  g1[2] = (int)(((tensor_x >> 16) & 0xffffu) | ((tensor_y & 0xffffu) << 16));
  g1[3] = (int)(((tensor_y >> 16) & 0xffffu) | ((tile_x & 0xffffu) << 16));
  g1[4] = (int)(tile_y & 0xffffu);                        // tile_dim1, tile_dim2=0
  g1[5] = (int)stride_elems;                              // tensor_dim0_stride[31:0]
  g1[6] = 0;
  g1[7] = 0;
  i32x4 z4 = {0, 0, 0, 0};
#if __clang_major__ >= 23
  i32x8 z8 = {0, 0, 0, 0, 0, 0, 0, 0};
  __builtin_amdgcn_tensor_load_to_lds(g0, g1, z4, z4, z8, 0);
#else
  __builtin_amdgcn_tensor_load_to_lds(g0, g1, z4, z4, 0);
#endif
}
#endif

// ---------------- f32 -> f16 convert ----------------
__global__ void cvt_kernel(const float* __restrict__ s, h16* __restrict__ d, int n) {
  int i = blockIdx.x * blockDim.x + threadIdx.x;
  if (i < n) d[i] = (h16)s[i];
}

// ---------------- GEMM: C = A(MxK) * W(NxK)^T ----------------
// Block: 256 thr = 8 waves; each wave: 16 rows x 64 cols. Grid: (M/128, N/64).
// A/W tiles staged in LDS: TDM double-buffered when available.
template <bool F32OUT>
__global__ __launch_bounds__(256, 1) void gemm_kernel(
    const h16* __restrict__ A, const h16* __restrict__ W,
    h16* __restrict__ Ch, float* __restrict__ Cf,
    const float* __restrict__ scaleP, int M, int N, int K) {
#ifdef HAVE_TDM
  __shared__ alignas(16) h16 At[2][128 * 32];   // 16 KB
  __shared__ alignas(16) h16 Wt[2][64 * 32];    //  8 KB
#else
  __shared__ alignas(16) h16 At[1][128 * 32];
  __shared__ alignas(16) h16 Wt[1][64 * 32];
#endif
  int lane = threadIdx.x & 31;
  int wave = threadIdx.x >> 5;
  int ln = lane & 15;
  int lh = (lane >> 4) & 1;
  int bm = blockIdx.x * 128;
  int n0 = blockIdx.y * 64;

  v8f zero = {};
  v8f acc[4];
  #pragma unroll
  for (int i = 0; i < 4; i++) acc[i] = zero;

#ifdef HAVE_TDM
  if (threadIdx.x == 0) {
    tdm_load_2d(A + (size_t)bm * K, &At[0][0], 32, 128, K, K, M);
    tdm_load_2d(W + (size_t)n0 * K, &Wt[0][0], 32, 64,  K, K, N);
  }
#endif
  for (int k0 = 0; k0 < K; k0 += 32) {
#ifdef HAVE_TDM
    int cur = (k0 >> 5) & 1;
    if (threadIdx.x == 0) {
      if (k0 + 32 < K) {
        tdm_load_2d(A + (size_t)bm * K + k0 + 32, &At[cur ^ 1][0], 32, 128, K, K, M);
        tdm_load_2d(W + (size_t)n0 * K + k0 + 32, &Wt[cur ^ 1][0], 32, 64,  K, K, N);
        __builtin_amdgcn_s_wait_tensorcnt(2);   // current pair complete (in-order)
      } else {
        __builtin_amdgcn_s_wait_tensorcnt(0);
      }
    }
    __syncthreads();
#else
    int cur = 0;
    __syncthreads();   // protect prior reads before overwrite
    {
      int r = threadIdx.x >> 1;
      int c = (threadIdx.x & 1) * 16;
      const h16* src = A + (size_t)(bm + r) * K + k0 + c;
      *(v8h*)&At[0][r * 32 + c]     = *(const v8h*)(src);
      *(v8h*)&At[0][r * 32 + c + 8] = *(const v8h*)(src + 8);
    }
    if (threadIdx.x < 64) {
      int r = threadIdx.x;
      const h16* src = W + (size_t)(n0 + r) * K + k0;
      #pragma unroll
      for (int i = 0; i < 4; i++)
        *(v8h*)&Wt[0][r * 32 + i * 8] = *(const v8h*)(src + i * 8);
    }
    __syncthreads();
#endif
    AF a;
    a.h[0] = *(const v8h*)&At[cur][(wave * 16 + ln) * 32 + lh * 16];
    a.h[1] = *(const v8h*)&At[cur][(wave * 16 + ln) * 32 + lh * 16 + 8];
    #pragma unroll
    for (int nt = 0; nt < 4; nt++) {
      AF bf;
      bf.h[0] = *(const v8h*)&Wt[cur][(nt * 16 + ln) * 32 + lh * 16];
      bf.h[1] = *(const v8h*)&Wt[cur][(nt * 16 + ln) * 32 + lh * 16 + 8];
      acc[nt] = wmma_f16(a.v, bf.v, acc[nt]);
    }
    __syncthreads();   // done reading before next stage overwrites
  }
  float scale = F32OUT ? (1.0f + scaleP[0]) : 1.0f;
  int m0 = bm + wave * 16;
  #pragma unroll
  for (int nt = 0; nt < 4; nt++) {
    #pragma unroll
    for (int r = 0; r < 8; r++) {
      size_t idx = (size_t)(m0 + r + lh * 8) * N + n0 + nt * 16 + ln;
      if (F32OUT) Cf[idx] = acc[nt][r] * scale;
      else        Ch[idx] = (h16)acc[nt][r];
    }
  }
}

// ---------------- K-shift + RoPE + RMSNorm + ve gate ----------------
__global__ __launch_bounds__(256, 1) void rope_kernel(
    h16* __restrict__ qh, const h16* __restrict__ khr, h16* __restrict__ vh,
    h16* __restrict__ khn,
    const float* __restrict__ x, const float* __restrict__ ve,
    const float* __restrict__ cosb, const float* __restrict__ sinb,
    const float* __restrict__ vgW) {
  int lane = threadIdx.x & 31;
  int wave = threadIdx.x >> 5;
  int idx = blockIdx.x * 8 + wave;
  int b = idx / (TLEN * NH);
  int rem = idx % (TLEN * NH);
  int t = rem / NH;
  int h = rem % NH;
  int j0 = lane * 2;
  size_t rowb = (size_t)(b * TLEN + t) * EMB;

  float c0 = cosb[t * 64 + j0], c1 = cosb[t * 64 + j0 + 1];
  float s0 = sinb[t * 64 + j0], s1 = sinb[t * 64 + j0 + 1];

  {
    h16* qp = qh + rowb + h * HD;
    float x1a = (float)qp[j0], x1b = (float)qp[j0 + 1];
    float x2a = (float)qp[j0 + 64], x2b = (float)qp[j0 + 65];
    float o1a = x1a * c0 + x2a * s0, o1b = x1b * c1 + x2b * s1;
    float o2a = -x1a * s0 + x2a * c0, o2b = -x1b * s1 + x2b * c1;
    float ss = o1a * o1a + o1b * o1b + o2a * o2a + o2b * o2b;
    for (int off = 1; off < 32; off <<= 1) ss += __shfl_xor(ss, off, 32);
    float r = rsqrtf(ss * (1.0f / HD) + 1.1920929e-07f);
    qp[j0] = (h16)(o1a * r); qp[j0 + 1] = (h16)(o1b * r);
    qp[j0 + 64] = (h16)(o2a * r); qp[j0 + 65] = (h16)(o2b * r);
  }
  {
    int tp = t > 0 ? t - 1 : 0;
    const h16* k1 = khr + rowb + h * HD;
    const h16* k2 = khr + (size_t)(b * TLEN + tp) * EMB + h * HD + 64;
    float x1a = (float)k1[j0], x1b = (float)k1[j0 + 1];
    float x2a = (float)k2[j0], x2b = (float)k2[j0 + 1];
    float o1a = x1a * c0 + x2a * s0, o1b = x1b * c1 + x2b * s1;
    float o2a = -x1a * s0 + x2a * c0, o2b = -x1b * s1 + x2b * c1;
    float ss = o1a * o1a + o1b * o1b + o2a * o2a + o2b * o2b;
    for (int off = 1; off < 32; off <<= 1) ss += __shfl_xor(ss, off, 32);
    float r = rsqrtf(ss * (1.0f / HD) + 1.1920929e-07f);
    h16* ko = khn + rowb + h * HD;
    ko[j0] = (h16)(o1a * r); ko[j0 + 1] = (h16)(o1b * r);
    ko[j0 + 64] = (h16)(o2a * r); ko[j0 + 65] = (h16)(o2b * r);
  }
  {
    float p = x[rowb + lane] * vgW[h * 32 + lane];
    for (int off = 1; off < 32; off <<= 1) p += __shfl_xor(p, off, 32);
    float gate = 2.0f / (1.0f + expf(-p));
    h16* vp = vh + rowb + h * HD;
    const float* vep = ve + rowb + h * HD;
    int ds4[4] = {j0, j0 + 1, j0 + 64, j0 + 65};
    #pragma unroll
    for (int i = 0; i < 4; i++) {
      int d = ds4[i];
      vp[d] = (h16)((float)vp[d] + gate * vep[d]);
    }
  }
}

// ---------------- Flash attention over paired heads ----------------
// Grid: (T/64, 8, B). Block 256 = 8 waves: 4 q-tiles x 2 column halves.
// Each wave: 16 queries x 128 d-columns -> acc is only 8 v8f (64 VGPRs).
__global__ __launch_bounds__(256, 1) void attn_kernel(
    const h16* __restrict__ qh, const h16* __restrict__ kh,
    const h16* __restrict__ vh,
    const float* __restrict__ x, const float* __restrict__ agW,
    h16* __restrict__ yh) {
  __shared__ alignas(16) h16 VT[256 * 40];        // V^T tile [d=256][k=32+pad]
  __shared__ alignas(16) h16 PL[4][16 * 40];      // per-qtile P [row][k+pad]
  __shared__ float SP[2][4][16 * 33];             // partial scores [half][qtile]

  int lane = threadIdx.x & 31;
  int wave = threadIdx.x >> 5;
  int qt = wave >> 1;         // 0..3
  int ch = wave & 1;          // column half: d in [ch*128, ch*128+128)
  int ln = lane & 15;
  int lh = (lane >> 4) & 1;
  int b = blockIdx.z, h2 = blockIdx.y;
  int q0b = blockIdx.x * 64;
  int qw = q0b + qt * 16;

  v8f zero = {};
  v8f acc[8];
  #pragma unroll
  for (int i = 0; i < 8; i++) acc[i] = zero;
  float mrow[8], lrow[8];
  #pragma unroll
  for (int r = 0; r < 8; r++) { mrow[r] = -3.0e38f; lrow[r] = 0.0f; }

  int t_lo = q0b - WIN; if (t_lo < 0) t_lo = 0; t_lo &= ~31;
  const h16* qbase = qh + (size_t)(b * TLEN + qw + ln) * EMB + h2 * D2;

  for (int t0 = t_lo; t0 < q0b + 64; t0 += 32) {
    __syncthreads();   // prior iteration finished with VT/SP
    // stage V^T into LDS (all 256 threads)
    {
      int k = threadIdx.x >> 3;
      int dbase = (threadIdx.x & 7) * 32;
      const h16* src = vh + (size_t)(b * TLEN + t0 + k) * EMB + h2 * D2 + dbase;
      #pragma unroll
      for (int i = 0; i < 4; i++) {
        v8h vv = *(const v8h*)(src + i * 8);
        #pragma unroll
        for (int e = 0; e < 8; e++) VT[(dbase + i * 8 + e) * 40 + k] = vv[e];
      }
    }
    // partial scores over this wave's 128 d-columns
    v8f S0 = zero, S1 = zero;
    #pragma unroll
    for (int dc = 0; dc < 4; dc++) {
      int d0 = (ch * 4 + dc) * 32;
      AF a;
      a.h[0] = *(const v8h*)(qbase + d0 + lh * 16);
      a.h[1] = *(const v8h*)(qbase + d0 + lh * 16 + 8);
      const h16* kb0 = kh + (size_t)(b * TLEN + t0 + ln) * EMB + h2 * D2 + d0 + lh * 16;
      const h16* kb1 = kb0 + (size_t)16 * EMB;
      AF b0, b1;
      b0.h[0] = *(const v8h*)(kb0); b0.h[1] = *(const v8h*)(kb0 + 8);
      b1.h[0] = *(const v8h*)(kb1); b1.h[1] = *(const v8h*)(kb1 + 8);
      S0 = wmma_f16(a.v, b0.v, S0);
      S1 = wmma_f16(a.v, b1.v, S1);
    }
    // exchange partials between the wave pair
    {
      float* sp = &SP[ch][qt][0];
      #pragma unroll
      for (int r = 0; r < 8; r++) {
        sp[(r + lh * 8) * 33 + ln]      = S0[r];
        sp[(r + lh * 8) * 33 + 16 + ln] = S1[r];
      }
    }
    __syncthreads();   // VT staged + partials visible
    {
      const float* spo = &SP[ch ^ 1][qt][0];
      #pragma unroll
      for (int r = 0; r < 8; r++) {
        S0[r] += spo[(r + lh * 8) * 33 + ln];
        S1[r] += spo[(r + lh * 8) * 33 + 16 + ln];
      }
    }
    // online softmax (computed identically in both waves of the pair)
    float p0[8], p1[8];
    #pragma unroll
    for (int r = 0; r < 8; r++) {
      int q = qw + r + lh * 8;
      int k0i = t0 + ln, k1i = t0 + 16 + ln;
      bool ok0 = (k0i <= q) && ((q - k0i) <= WIN);
      bool ok1 = (k1i <= q) && ((q - k1i) <= WIN);
      float s0v = ok0 ? S0[r] * 0.0625f : -3.0e38f;
      float s1v = ok1 ? S1[r] * 0.0625f : -3.0e38f;
      float v = fmaxf(s0v, s1v);
      for (int off = 1; off < 16; off <<= 1) v = fmaxf(v, __shfl_xor(v, off, 32));
      float mnew = fmaxf(mrow[r], v);
      float sc = expf(mrow[r] - mnew);
      float e0 = ok0 ? expf(s0v - mnew) : 0.0f;
      float e1 = ok1 ? expf(s1v - mnew) : 0.0f;
      float rs = e0 + e1;
      for (int off = 1; off < 16; off <<= 1) rs += __shfl_xor(rs, off, 32);
      lrow[r] = lrow[r] * sc + rs;
      mrow[r] = mnew;
      p0[r] = e0; p1[r] = e1;
      #pragma unroll
      for (int i = 0; i < 8; i++) acc[i][r] *= sc;
    }
    // both waves write identical P tile; each reads only its own writes
    #pragma unroll
    for (int r = 0; r < 8; r++) {
      PL[qt][(r + lh * 8) * 40 + ln]      = (h16)p0[r];
      PL[qt][(r + lh * 8) * 40 + 16 + ln] = (h16)p1[r];
    }
    AF pa;
    const h16* pb = &PL[qt][ln * 40 + lh * 16];
    pa.h[0] = *(const v8h*)(pb);
    pa.h[1] = *(const v8h*)(pb + 8);
    // O += P(16x32) . V(32x128) on this wave's column half
    #pragma unroll
    for (int dc = 0; dc < 8; dc++) {
      int d16 = ch * 8 + dc;
      const h16* vbp = &VT[(d16 * 16 + ln) * 40 + lh * 16];
      AF vb;
      vb.h[0] = *(const v8h*)(vbp);
      vb.h[1] = *(const v8h*)(vbp + 8);
      acc[dc] = wmma_f16(pa.v, vb.v, acc[dc]);
    }
  }

  // epilogue: 1/l, attention gate (this wave covers original head 2*h2+ch)
  #pragma unroll
  for (int r = 0; r < 8; r++) {
    int q = qw + r + lh * 8;
    float li = 1.0f / lrow[r];
    const float* xr = x + (size_t)(b * TLEN + q) * EMB;
    float g = 0.0f;
    #pragma unroll
    for (int c = 0; c < 12; c++) g += xr[c] * agW[(2 * h2 + ch) * 12 + c];
    g = 1.0f / (1.0f + expf(-g));
    h16* yrow = yh + (size_t)(b * TLEN + q) * EMB + h2 * D2 + ch * 128;
    #pragma unroll
    for (int dc = 0; dc < 8; dc++)
      yrow[dc * 16 + ln] = (h16)(acc[dc][r] * li * g);
  }
}

// ---------------- host ----------------
extern "C" void kernel_launch(void* const* d_in, const int* in_sizes, int n_in,
                              void* d_out, int out_size, void* d_ws, size_t ws_size,
                              hipStream_t stream) {
  (void)in_sizes; (void)n_in; (void)out_size; (void)ws_size;
  const float* x    = (const float*)d_in[0];
  const float* ve   = (const float*)d_in[1];
  const float* cosb = (const float*)d_in[2];
  const float* sinb = (const float*)d_in[3];
  const float* Wq   = (const float*)d_in[4];
  const float* Wk   = (const float*)d_in[5];
  const float* Wv   = (const float*)d_in[6];
  const float* Wo   = (const float*)d_in[7];
  const float* cps  = (const float*)d_in[8];
  const float* vgW  = (const float*)d_in[9];
  const float* agW  = (const float*)d_in[10];
  float* out = (float*)d_out;

  char* ws = (char*)d_ws;
  const size_t MB = (size_t)1 << 20;
  h16* xh  = (h16*)(ws + 0 * MB);
  h16* wqh = (h16*)(ws + 16 * MB);
  h16* wkh = (h16*)(ws + 24 * MB);
  h16* wvh = (h16*)(ws + 32 * MB);
  h16* woh = (h16*)(ws + 40 * MB);
  h16* qh  = (h16*)(ws + 48 * MB);
  h16* khr = (h16*)(ws + 64 * MB);
  h16* vhb = (h16*)(ws + 80 * MB);
  h16* khn = (h16*)(ws + 96 * MB);
  h16* yh  = (h16*)(ws + 112 * MB);

  int nx = BT * EMB;
  int nw = EMB * EMB;
  cvt_kernel<<<nx / 256, 256, 0, stream>>>(x,  xh,  nx);
  cvt_kernel<<<nw / 256, 256, 0, stream>>>(Wq, wqh, nw);
  cvt_kernel<<<nw / 256, 256, 0, stream>>>(Wk, wkh, nw);
  cvt_kernel<<<nw / 256, 256, 0, stream>>>(Wv, wvh, nw);
  cvt_kernel<<<nw / 256, 256, 0, stream>>>(Wo, woh, nw);

  dim3 gg(BT / 128, EMB / 64);
  gemm_kernel<false><<<gg, 256, 0, stream>>>(xh, wqh, qh,  nullptr, nullptr, BT, EMB, EMB);
  gemm_kernel<false><<<gg, 256, 0, stream>>>(xh, wkh, khr, nullptr, nullptr, BT, EMB, EMB);
  gemm_kernel<false><<<gg, 256, 0, stream>>>(xh, wvh, vhb, nullptr, nullptr, BT, EMB, EMB);

  rope_kernel<<<BT * NH / 8, 256, 0, stream>>>(qh, khr, vhb, khn, x, ve, cosb, sinb, vgW);

  attn_kernel<<<dim3(TLEN / 64, H2G, BDIM), 256, 0, stream>>>(qh, khn, vhb, x, agW, yh);

  gemm_kernel<true><<<gg, 256, 0, stream>>>(yh, woh, nullptr, out, cps, BT, EMB, EMB);
}
